// FinancialMixtureOfExperts_15109694948208
// MI455X (gfx1250) — matrix operand order repaired and checked by reading, not compiled, not run on hardware
//
#include <hip/hip_runtime.h>
#include <hip/hip_bf16.h>

// ---------------------------------------------------------------------------
// MoE transformer forward on gfx1250 (MI455X), bf16 WMMA everywhere.
// GEMMs: v_wmma_f32_16x16x32_bf16, 128x64 block tile, 2x2 WMMA per wave.
// Staging: GLOBAL_LOAD_ASYNC_TO_LDS_B128 (ASYNCcnt) when available.
// ---------------------------------------------------------------------------

typedef __bf16 bf16;
typedef __attribute__((ext_vector_type(16))) __bf16 v16bf;
typedef __attribute__((ext_vector_type(8)))  float  v8f;

// Problem constants (from reference)
#define CB   512      // batch
#define CS   64       // seq
#define CIN  128      // input dim
#define CDM  256      // model dim
#define CH   8        // heads
#define CL   2        // layers
#define CFF  1024     // ff dim
#define CDO  64       // out dim
#define CE   8        // experts
#define CHD  32       // head dim
#define CGI  8192     // S*IN gate input
#define CT   32768    // B*S tokens

// ---------------------------------------------------------------------------
// Async global->LDS staging (CDNA5 GLOBAL_LOAD_ASYNC_TO_LDS_B128, ASYNCcnt).
// Builtin signature (from compiler diagnostic): param0 = addrspace(1) v4i*,
// param1 = addrspace(3) v4i*, then imm offset + imm cpol.
// Guarded: falls back to VGPR round-trip copy if the builtin is absent.
// ---------------------------------------------------------------------------
#if defined(__gfx1250__) && __has_builtin(__builtin_amdgcn_global_load_async_to_lds_b128)
#define ASYNC_LDS 1
typedef int v4i_gcc __attribute__((vector_size(16)));
typedef __attribute__((address_space(1))) void    v_as1;
typedef __attribute__((address_space(3))) void    v_as3;
typedef __attribute__((address_space(1))) v4i_gcc v4i_as1;
typedef __attribute__((address_space(3))) v4i_gcc v4i_as3;
#endif

__device__ __forceinline__ void async_copy16(const bf16* gsrc, bf16* lds) {
#ifdef ASYNC_LDS
    __builtin_amdgcn_global_load_async_to_lds_b128(
        (v4i_as1*)(v_as1*)gsrc, (v4i_as3*)(v_as3*)lds, 0, 0);
#else
    *(uint4*)lds = *(const uint4*)gsrc;
#endif
}

__device__ __forceinline__ void wait_async() {
#ifdef ASYNC_LDS
#if __has_builtin(__builtin_amdgcn_s_wait_asynccnt)
    __builtin_amdgcn_s_wait_asynccnt(0);
#else
    asm volatile("s_wait_asynccnt 0x0" ::: "memory");
#endif
#endif
}

// ---------------------------------------------------------------------------
// f32 -> bf16 cast
// ---------------------------------------------------------------------------
__global__ void cast_kernel(const float* __restrict__ in, bf16* __restrict__ out, int n) {
    int i = blockIdx.x * 256 + threadIdx.x;
    if (i < n) out[i] = (bf16)in[i];
}

// ---------------------------------------------------------------------------
// Generic GEMM: C[M,N] = act(A[M,K] @ W[K,N] + bias), A/W bf16, f32 acc.
// Block = 256 thr = 8 waves arranged 4(M) x 2(N); block tile 128 x 64.
// Each wave owns a 32x32 output tile = 2x2 WMMA tiles -> 4 WMMA per K-step,
// with A-fragment reuse across N and B-fragment reuse across M.
// ---------------------------------------------------------------------------
__global__ __launch_bounds__(256) void gemm_bf16(
    const bf16* __restrict__ A, const bf16* __restrict__ W,
    const float* __restrict__ bias, float* __restrict__ Cf,
    bf16* __restrict__ Cb, int M, int N, int Kdim, int relu)
{
    __shared__ bf16 lA[128][32];   // A tile, row-major          (8 KB)
    __shared__ bf16 lWt[64][32];   // W tile transposed: [n][k]  (4 KB)

    const int t    = threadIdx.x;
    const int lane = t & 31;
    const int wave = t >> 5;
    const int wm   = wave >> 1;         // 0..3  (M subtile of 32)
    const int wn   = wave & 1;          // 0..1  (N subtile of 32)
    const int mBlock = blockIdx.y * 128;
    const int nBlock = blockIdx.x * 64;

    v8f c00 = {}, c01 = {}, c10 = {}, c11 = {};

    // staging indices: A = 128x32 (4096 el) -> 16 el (32B) per thread
    const int arowS = t >> 1;
    const int asegS = (t & 1) * 16;

    for (int k0 = 0; k0 < Kdim; k0 += 32) {
        __syncthreads();
        // ---- stage A tile (async global->LDS, no VGPR round trip) ----
        {
            const bf16* src = A + (size_t)(mBlock + arowS) * Kdim + k0 + asegS;
            bf16* dst = &lA[arowS][asegS];
            async_copy16(src,     dst);
            async_copy16(src + 8, dst + 8);
            if (k0 + 32 < Kdim) __builtin_prefetch(src + 32, 0, 1);
        }
        // ---- stage W tile transposed: lWt[n][k] = W[k0+k][nBlock+n] ----
        #pragma unroll
        for (int j = 0; j < 8; ++j) {
            int idx = t * 8 + j;
            int kk = idx >> 6, n = idx & 63;
            lWt[n][kk] = W[(size_t)(k0 + kk) * N + nBlock + n];
        }
        wait_async();
        __syncthreads();

        // ---- fragments (16-bit layouts, ISA 7.12.2) ----
        v16bf a0, a1, b0, b1;
        const int ar0 = wm * 32 + (lane & 15);
        const int ar1 = ar0 + 16;
        const int akb = (lane < 16) ? 0 : 8;
        #pragma unroll
        for (int j = 0; j < 8; ++j) {
            a0[j]     = lA[ar0][akb + j];
            a0[8 + j] = lA[ar0][akb + 16 + j];
            a1[j]     = lA[ar1][akb + j];
            a1[8 + j] = lA[ar1][akb + 16 + j];
        }
        const int bc0 = wn * 32 + (lane & 15);
        const int bc1 = bc0 + 16;
        const int bkb = (lane < 16) ? 0 : 8;
        #pragma unroll
        for (int j = 0; j < 8; ++j) { b0[j] = lWt[bc0][bkb + j]; b1[j] = lWt[bc1][bkb + j]; b0[8 + j] = lWt[bc0][bkb + 16 + j]; b1[8 + j] = lWt[bc1][bkb + 16 + j]; }

        c00 = __builtin_amdgcn_wmma_f32_16x16x32_bf16(false, a0, false, b0, (short)0, c00, false, false);
        c01 = __builtin_amdgcn_wmma_f32_16x16x32_bf16(false, a0, false, b1, (short)0, c01, false, false);
        c10 = __builtin_amdgcn_wmma_f32_16x16x32_bf16(false, a1, false, b0, (short)0, c10, false, false);
        c11 = __builtin_amdgcn_wmma_f32_16x16x32_bf16(false, a1, false, b1, (short)0, c11, false, false);
    }

    // ---- epilogue: C/D layout -> (M = r + 8*(lane>=16), N = lane%16) ----
    const int mloc = (lane < 16) ? 0 : 8;
    const v8f* cc[2][2] = { { &c00, &c01 }, { &c10, &c11 } };
    #pragma unroll
    for (int ti = 0; ti < 2; ++ti) {
        #pragma unroll
        for (int tj = 0; tj < 2; ++tj) {
            const int gc = nBlock + wn * 32 + tj * 16 + (lane & 15);
            const float bvv = bias ? bias[gc] : 0.0f;
            #pragma unroll
            for (int r = 0; r < 8; ++r) {
                const int grow = mBlock + wm * 32 + ti * 16 + mloc + r;
                float v = (*cc[ti][tj])[r] + bvv;
                if (relu) v = fmaxf(v, 0.0f);
                if (Cf) Cf[(size_t)grow * N + gc] = v;
                if (Cb) Cb[(size_t)grow * N + gc] = (bf16)v;
            }
        }
    }
}

// ---------------------------------------------------------------------------
// Fused attention per (b,h): scores = q k^T * scale -> softmax -> attn @ v
// q/k/v/av layout: [B, S, H, HD] (== [B,S,DM] head slice), bf16.
// One block per (b,h); 8 waves. S=64, HD=32 (one WMMA K-step for scores).
// ---------------------------------------------------------------------------
__global__ __launch_bounds__(256) void attn_kernel(
    const bf16* __restrict__ Q, const bf16* __restrict__ Kt,
    const bf16* __restrict__ V, bf16* __restrict__ AV, float scale)
{
    __shared__ bf16  lq[64][32];
    __shared__ bf16  lk[64][32];
    __shared__ bf16  lvT[32][64];
    __shared__ float lsc[64][64];
    __shared__ bf16  lat[64][64];

    const int t = threadIdx.x, lane = t & 31, wave = t >> 5;
    const int bh = blockIdx.x;
    const int b  = bh >> 3;      // /H
    const int h  = bh & 7;       // %H
    const size_t base = (size_t)b * CS * CDM + (size_t)h * CHD;

    // Stage q, k via async global->LDS (16B per thread each)
    {
        const int row = t >> 2, seg = (t & 3) * 8;
        const size_t off = base + (size_t)row * CDM + seg;
        async_copy16(&Q[off],  &lq[row][seg]);
        async_copy16(&Kt[off], &lk[row][seg]);
    }
    // Stage v transposed: lvT[d][s] (scalar; transpose can't go async)
    #pragma unroll
    for (int j = 0; j < 8; ++j) {
        int idx = t * 8 + j;
        int row = idx >> 5, d = idx & 31;
        lvT[d][row] = V[base + (size_t)row * CDM + d];
    }
    wait_async();
    __syncthreads();

    // Scores: 16 tiles of 16x16 (4x4), each wave does 2 tiles; K=HD=32 -> 1 WMMA
    for (int tt = wave; tt < 16; tt += 8) {
        const int ti = tt >> 2, tj = tt & 3;
        v16bf a, bb; v8f c = {};
        const int arow = ti * 16 + (lane & 15);
        const int akb  = (lane < 16) ? 0 : 8;
        #pragma unroll
        for (int j = 0; j < 8; ++j) {
            a[j]     = lq[arow][akb + j];
            a[8 + j] = lq[arow][akb + 16 + j];
        }
        const int bcol = tj * 16 + (lane & 15);
        const int bkb  = (lane < 16) ? 0 : 8;
        #pragma unroll
        for (int j = 0; j < 8; ++j) { bb[j] = lk[bcol][bkb + j]; bb[8 + j] = lk[bcol][bkb + 16 + j]; }
        c = __builtin_amdgcn_wmma_f32_16x16x32_bf16(
            false, a, false, bb, (short)0, c, false, false);
        const int mloc = (lane < 16) ? 0 : 8;
        #pragma unroll
        for (int r = 0; r < 8; ++r)
            lsc[ti * 16 + mloc + r][tj * 16 + (lane & 15)] = c[r] * scale;
    }
    __syncthreads();

    // Softmax rows (f32 in LDS), emit bf16 attn
    if (t < 64) {
        float mx = -1e30f;
        for (int j = 0; j < 64; ++j) mx = fmaxf(mx, lsc[t][j]);
        float sum = 0.0f;
        for (int j = 0; j < 64; ++j) { float e = expf(lsc[t][j] - mx); lsc[t][j] = e; sum += e; }
        const float inv = 1.0f / sum;
        for (int j = 0; j < 64; ++j) lat[t][j] = (bf16)(lsc[t][j] * inv);
    }
    __syncthreads();

    // av = attn(64x64) @ v(64x32): 8 tiles (4x2), one per wave, K=64 -> 2 WMMA
    {
        const int ti = wave >> 1, tj = wave & 1;
        v8f c = {};
        #pragma unroll
        for (int k0 = 0; k0 < 64; k0 += 32) {
            v16bf a, bb;
            const int arow = ti * 16 + (lane & 15);
            const int akb  = k0 + ((lane < 16) ? 0 : 8);
            #pragma unroll
            for (int j = 0; j < 8; ++j) {
                a[j]     = lat[arow][akb + j];
                a[8 + j] = lat[arow][akb + 16 + j];
            }
            const int bcol = tj * 16 + (lane & 15);
            const int bkb  = k0 + ((lane < 16) ? 0 : 8);
            #pragma unroll
            for (int j = 0; j < 8; ++j) { bb[j] = lvT[bcol][bkb + j]; bb[8 + j] = lvT[bcol][bkb + 16 + j]; }
            c = __builtin_amdgcn_wmma_f32_16x16x32_bf16(
                false, a, false, bb, (short)0, c, false, false);
        }
        const int mloc = (lane < 16) ? 0 : 8;
        const int dcol = tj * 16 + (lane & 15);
        #pragma unroll
        for (int r = 0; r < 8; ++r) {
            const int srow = ti * 16 + mloc + r;
            AV[base + (size_t)srow * CDM + dcol] = (bf16)c[r];
        }
    }
}

// ---------------------------------------------------------------------------
// Gate: logits = gi @ Wg + bg; top-2 softmax -> combine weights cw[B,E]
// ---------------------------------------------------------------------------
__global__ __launch_bounds__(256) void gate_kernel(
    const float* __restrict__ x, const float* __restrict__ Wg,
    const float* __restrict__ bg, float* __restrict__ cw)
{
    __shared__ float part[256][CE];
    const int t = threadIdx.x, b = blockIdx.x;
    float p[CE];
    #pragma unroll
    for (int e = 0; e < CE; ++e) p[e] = 0.0f;
    const float* gi = x + (size_t)b * CGI;
    for (int i = t; i < CGI; i += 256) {
        const float g = gi[i];
        const float* wrow = Wg + (size_t)i * CE;
        #pragma unroll
        for (int e = 0; e < CE; ++e) p[e] += g * wrow[e];
    }
    #pragma unroll
    for (int e = 0; e < CE; ++e) part[t][e] = p[e];
    __syncthreads();
    for (int s = 128; s > 0; s >>= 1) {
        if (t < s) {
            #pragma unroll
            for (int e = 0; e < CE; ++e) part[t][e] += part[t + s][e];
        }
        __syncthreads();
    }
    if (t == 0) {
        float logits[CE];
        #pragma unroll
        for (int e = 0; e < CE; ++e) logits[e] = part[0][e] + bg[e];
        int i1 = 0;
        for (int e = 1; e < CE; ++e) if (logits[e] > logits[i1]) i1 = e;
        int i2 = -1;
        for (int e = 0; e < CE; ++e)
            if (e != i1 && (i2 < 0 || logits[e] > logits[i2])) i2 = e;
        const float m  = logits[i1];
        const float e1 = expf(logits[i1] - m), e2 = expf(logits[i2] - m);
        const float inv = 1.0f / (e1 + e2);
        for (int e = 0; e < CE; ++e) cw[b * CE + e] = 0.0f;
        cw[b * CE + i1] = e1 * inv;
        cw[b * CE + i2] = e2 * inv;
    }
}

// ---------------------------------------------------------------------------
// h = LN(h + proj) * g + b; also emit bf16 copy.  One block per token (DM=256).
// ---------------------------------------------------------------------------
__global__ __launch_bounds__(256) void add_ln_kernel(
    float* __restrict__ hio, const float* __restrict__ proj,
    const float* __restrict__ g, const float* __restrict__ bb,
    bf16* __restrict__ hb)
{
    __shared__ float red[256];
    const int t = threadIdx.x;
    const size_t row = blockIdx.x;
    const float v = hio[row * CDM + t] + proj[row * CDM + t];
    red[t] = v; __syncthreads();
    for (int s = 128; s > 0; s >>= 1) { if (t < s) red[t] += red[t + s]; __syncthreads(); }
    const float mean = red[0] * (1.0f / CDM); __syncthreads();
    const float d = v - mean;
    red[t] = d * d; __syncthreads();
    for (int s = 128; s > 0; s >>= 1) { if (t < s) red[t] += red[t + s]; __syncthreads(); }
    const float var = red[0] * (1.0f / CDM);
    const float o = d * rsqrtf(var + 1e-5f) * g[t] + bb[t];
    hio[row * CDM + t] = o;
    hb[row * CDM + t]  = (bf16)o;
}

// hmean[b,d] = mean_s h[b,s,d] (bf16 out for final GEMM)
__global__ void mean_kernel(const float* __restrict__ h, bf16* __restrict__ hm) {
    const int id = blockIdx.x * 256 + threadIdx.x;   // B*DM
    const int b = id >> 8, d = id & 255;
    float s = 0.0f;
    for (int ss = 0; ss < CS; ++ss) s += h[((size_t)(b * CS + ss)) * CDM + d];
    hm[id] = (bf16)(s * (1.0f / CS));
}

__global__ void zero_kernel(float* __restrict__ p, int n) {
    const int i = blockIdx.x * 256 + threadIdx.x;
    if (i < n) p[i] = 0.0f;
}

// acc[b,d] += cw[b,e] * eo[b,d]
__global__ void accum_kernel(float* __restrict__ acc, const float* __restrict__ eo,
                             const float* __restrict__ cw, int e) {
    const int id = blockIdx.x * 256 + threadIdx.x;   // B*DO
    const int b = id >> 6;
    acc[id] += cw[b * CE + e] * eo[id];
}

// out = LN(acc + 0.1*r) * on_g + on_b  (one block per batch row, DO=64)
__global__ __launch_bounds__(64) void final_kernel(
    const float* __restrict__ acc, const float* __restrict__ rbuf,
    const float* __restrict__ og, const float* __restrict__ ob,
    float* __restrict__ out)
{
    __shared__ float red[64];
    const int t = threadIdx.x, b = blockIdx.x;
    const float v = acc[b * CDO + t] + 0.1f * rbuf[b * CDO + t];
    red[t] = v; __syncthreads();
    for (int s = 32; s > 0; s >>= 1) { if (t < s) red[t] += red[t + s]; __syncthreads(); }
    const float mean = red[0] * (1.0f / CDO); __syncthreads();
    const float d = v - mean;
    red[t] = d * d; __syncthreads();
    for (int s = 32; s > 0; s >>= 1) { if (t < s) red[t] += red[t + s]; __syncthreads(); }
    const float var = red[0] * (1.0f / CDO);
    out[b * CDO + t] = d * rsqrtf(var + 1e-5f) * og[t] + ob[t];
}

// ---------------------------------------------------------------------------
// Host orchestration
// ---------------------------------------------------------------------------
extern "C" void kernel_launch(void* const* d_in, const int* in_sizes, int n_in,
                              void* d_out, int out_size, void* d_ws, size_t ws_size,
                              hipStream_t stream) {
    (void)in_sizes; (void)n_in; (void)out_size; (void)ws_size;
    const float* x    = (const float*)d_in[0];
    const float* Win  = (const float*)d_in[1];
    const float* bin_ = (const float*)d_in[2];
    const float* Wq   = (const float*)d_in[3];
    const float* bq   = (const float*)d_in[4];
    const float* Wk   = (const float*)d_in[5];
    const float* bk   = (const float*)d_in[6];
    const float* Wv   = (const float*)d_in[7];
    const float* bv   = (const float*)d_in[8];
    const float* Wo   = (const float*)d_in[9];
    const float* bo   = (const float*)d_in[10];
    const float* ln1g = (const float*)d_in[11];
    const float* ln1b = (const float*)d_in[12];
    const float* ln2g = (const float*)d_in[13];
    const float* ln2b = (const float*)d_in[14];
    const float* W1   = (const float*)d_in[15];
    const float* b1   = (const float*)d_in[16];
    const float* W2   = (const float*)d_in[17];
    const float* b2   = (const float*)d_in[18];
    const float* Wout = (const float*)d_in[19];
    const float* bout = (const float*)d_in[20];
    const float* Wg   = (const float*)d_in[21];
    const float* bg   = (const float*)d_in[22];
    const float* Wr   = (const float*)d_in[23];
    const float* br   = (const float*)d_in[24];
    const float* ong  = (const float*)d_in[25];
    const float* onb  = (const float*)d_in[26];

    // Workspace layout (~255 MB)
    char* ws = (char*)d_ws;
    size_t off = 0;
    auto alloc = [&](size_t bytes) -> void* {
        off = (off + 255) & ~(size_t)255;
        void* p = ws + off; off += bytes; return p;
    };
    float* cw     = (float*)alloc((size_t)CB * CE * 4);
    bf16*  xb     = (bf16*) alloc((size_t)CT * CIN * 2);
    bf16*  Win_b  = (bf16*) alloc((size_t)CE * CIN * CDM * 2);
    bf16*  Wq_b   = (bf16*) alloc((size_t)CE * CL * CDM * CDM * 2);
    bf16*  Wk_b   = (bf16*) alloc((size_t)CE * CL * CDM * CDM * 2);
    bf16*  Wv_b   = (bf16*) alloc((size_t)CE * CL * CDM * CDM * 2);
    bf16*  Wo_b   = (bf16*) alloc((size_t)CE * CL * CDM * CDM * 2);
    bf16*  W1_b   = (bf16*) alloc((size_t)CE * CL * CDM * CFF * 2);
    bf16*  W2_b   = (bf16*) alloc((size_t)CE * CL * CFF * CDM * 2);
    bf16*  Wout_b = (bf16*) alloc((size_t)CE * CDM * CDO * 2);
    bf16*  Wr_b   = (bf16*) alloc((size_t)CGI * CDO * 2);
    float* h      = (float*)alloc((size_t)CT * CDM * 4);
    bf16*  hb     = (bf16*) alloc((size_t)CT * CDM * 2);
    bf16*  qb     = (bf16*) alloc((size_t)CT * CDM * 2);
    bf16*  kb     = (bf16*) alloc((size_t)CT * CDM * 2);
    bf16*  vb     = (bf16*) alloc((size_t)CT * CDM * 2);
    bf16*  avb    = (bf16*) alloc((size_t)CT * CDM * 2);
    float* proj   = (float*)alloc((size_t)CT * CDM * 4);
    bf16*  ffb    = (bf16*) alloc((size_t)CT * CFF * 2);
    bf16*  hm     = (bf16*) alloc((size_t)CB * CDM * 2);
    float* eo     = (float*)alloc((size_t)CB * CDO * 4);
    float* acc    = (float*)alloc((size_t)CB * CDO * 4);
    float* rbuf   = (float*)alloc((size_t)CB * CDO * 4);

    auto cast = [&](const float* src, bf16* dst, size_t n) {
        cast_kernel<<<(unsigned)((n + 255) / 256), 256, 0, stream>>>(src, dst, (int)n);
    };
    auto gemm = [&](const bf16* A, const bf16* W, const float* bias, float* Cf,
                    bf16* Cb, int M, int N, int Kd, int relu) {
        dim3 g((unsigned)(N / 64), (unsigned)(M / 128));
        gemm_bf16<<<g, 256, 0, stream>>>(A, W, bias, Cf, Cb, M, N, Kd, relu);
    };

    // Casts
    cast(x,    xb,     (size_t)CT * CIN);
    cast(Win,  Win_b,  (size_t)CE * CIN * CDM);
    cast(Wq,   Wq_b,   (size_t)CE * CL * CDM * CDM);
    cast(Wk,   Wk_b,   (size_t)CE * CL * CDM * CDM);
    cast(Wv,   Wv_b,   (size_t)CE * CL * CDM * CDM);
    cast(Wo,   Wo_b,   (size_t)CE * CL * CDM * CDM);
    cast(W1,   W1_b,   (size_t)CE * CL * CDM * CFF);
    cast(W2,   W2_b,   (size_t)CE * CL * CFF * CDM);
    cast(Wout, Wout_b, (size_t)CE * CDM * CDO);
    cast(Wr,   Wr_b,   (size_t)CGI * CDO);

    // Gate + accumulator init
    gate_kernel<<<CB, 256, 0, stream>>>(x, Wg, bg, cw);
    zero_kernel<<<(CB * CDO) / 256, 256, 0, stream>>>(acc, CB * CDO);

    const float scale = 0.17677669529663689f;  // HD^-0.5

    for (int e = 0; e < CE; ++e) {
        // h = x @ Win[e] + bin_[e]
        gemm(xb, Win_b + (size_t)e * CIN * CDM, bin_ + (size_t)e * CDM,
             h, hb, CT, CDM, CIN, 0);

        for (int l = 0; l < CL; ++l) {
            const size_t wo  = (size_t)(e * CL + l) * CDM * CDM;
            const size_t bo_ = (size_t)(e * CL + l) * CDM;
            // q, k, v projections (bf16 outputs for attention)
            gemm(hb, Wq_b + wo, bq + bo_, nullptr, qb, CT, CDM, CDM, 0);
            gemm(hb, Wk_b + wo, bk + bo_, nullptr, kb, CT, CDM, CDM, 0);
            gemm(hb, Wv_b + wo, bv + bo_, nullptr, vb, CT, CDM, CDM, 0);
            // attention
            attn_kernel<<<CB * CH, 256, 0, stream>>>(qb, kb, vb, avb, scale);
            // output projection + residual LN1
            gemm(avb, Wo_b + wo, bo + bo_, proj, nullptr, CT, CDM, CDM, 0);
            add_ln_kernel<<<CT, 256, 0, stream>>>(
                h, proj, ln1g + bo_, ln1b + bo_, hb);
            // FF: relu(h @ W1 + b1) @ W2 + b2, residual LN2
            const size_t w1o = (size_t)(e * CL + l) * CDM * CFF;
            const size_t b1o = (size_t)(e * CL + l) * CFF;
            gemm(hb, W1_b + w1o, b1 + b1o, nullptr, ffb, CT, CFF, CDM, 1);
            gemm(ffb, W2_b + w1o, b2 + bo_, proj, nullptr, CT, CDM, CFF, 0);
            add_ln_kernel<<<CT, 256, 0, stream>>>(
                h, proj, ln2g + bo_, ln2b + bo_, hb);
        }
        // expert head: mean over S, project to DO, weighted combine
        mean_kernel<<<(CB * CDM) / 256, 256, 0, stream>>>(h, hm);
        gemm(hm, Wout_b + (size_t)e * CDM * CDO, bout + (size_t)e * CDO,
             eo, nullptr, CB, CDO, CDM, 0);
        accum_kernel<<<(CB * CDO) / 256, 256, 0, stream>>>(acc, eo, cw, e);
    }

    // Residual projection: r = gi @ Wr + br ; out = LN(acc + 0.1*r)
    gemm(xb, Wr_b, br, rbuf, nullptr, CB, CDO, CGI, 0);
    final_kernel<<<CB, 64, 0, stream>>>(acc, rbuf, ong, onb, (float*)d_out);
}
